// GraphRNNDecoder_71734543778145
// MI455X (gfx1250) — matrix operand order repaired and checked by reading, not compiled, not run on hardware
//
#include <hip/hip_runtime.h>

typedef __attribute__((ext_vector_type(16))) _Float16 v16h;
typedef __attribute__((ext_vector_type(8)))  _Float16 v8h;
typedef __attribute__((ext_vector_type(8)))  float    v8f;

#define BATCH  4096
#define ZDIM   256
#define HID    512
#define FEAT   128
#define TSTEPS 128

// ---------------------------------------------------------------- helpers

__device__ __forceinline__ float sigmoid_f(float x) {
    return 1.0f / (1.0f + __expf(-x));
}
__device__ __forceinline__ float tanh_f(float x) {
    // 1 - 2/(e^{2x}+1): branchless, saturates correctly to +/-1
    return 1.0f - 2.0f / (__expf(2.0f * x) + 1.0f);
}

// A-fragment (16x32 f16, contiguous k order, same as B): lane holds row M=lane&15.
// element e <-> K = k0 + 16*half + e
__device__ __forceinline__ v16h load_a16(const _Float16* __restrict__ rowp,
                                         int k0, int half) {
    v8h lo = *(const v8h*)(rowp + k0 + 16 * half);
    v8h hi = *(const v8h*)(rowp + k0 + 16 * half + 8);
    return __builtin_shufflevector(lo, hi, 0, 1, 2, 3, 4, 5, 6, 7,
                                          8, 9, 10, 11, 12, 13, 14, 15);
}

// B-fragment (32x16 f16): lane holds col N=lane&15, element e <-> K = k0+16*half+e
// => one contiguous 32-byte load from the weight row (weights stored [N, K] row-major).
#define LOAD_B(Wrow, kb) (*(const v16h*)((Wrow) + (kb)))

#define WMMA(acc, a, b)                                                        \
    acc = __builtin_amdgcn_wmma_f32_16x16x32_f16(false, (a), false, (b),       \
                                                 (short)0, (acc), false, false)

// ---------------------------------------------------------------- cvt f32->f16

__global__ void cvt_f32_f16(const float* __restrict__ in,
                            _Float16* __restrict__ out, int n) {
    int i = blockIdx.x * blockDim.x + threadIdx.x;
    if (i < n) out[i] = (_Float16)in[i];
}

// ---------------------------------------------------------------- h0 = relu(z @ Winit^T + b)
// grid: (HID/32, BATCH/64), block 256 (8 waves, 4 row-slices x 2 col-slices)

__global__ __launch_bounds__(256) void init_h0(
    const _Float16* __restrict__ z16,      // [BATCH, ZDIM]
    const _Float16* __restrict__ Winit16,  // [HID, ZDIM] row-major
    const float* __restrict__ binit,       // [HID]
    float* __restrict__ h32,               // [BATCH, HID]
    _Float16* __restrict__ h16) {          // [BATCH, HID]
    const int lane = threadIdx.x & 31;
    const int w = threadIdx.x >> 5;
    const int wm = w & 3, wn = w >> 2;
    const int m = lane & 15, half = lane >> 4;

    const int row0 = blockIdx.y * 64 + wm * 16;
    const int col0 = blockIdx.x * 32 + wn * 16;
    const int j = col0 + m;

    const _Float16* Arow = z16 + (size_t)(row0 + m) * ZDIM;
    const _Float16* Brow = Winit16 + (size_t)j * ZDIM;

    v8f acc = {};
    for (int k0 = 0; k0 < ZDIM; k0 += 32) {
        v16h a = load_a16(Arow, k0, half);
        v16h b = LOAD_B(Brow, k0 + 16 * half);
        WMMA(acc, a, b);
    }

    const float bj = binit[j];
#pragma unroll
    for (int r = 0; r < 8; ++r) {
        int row = row0 + r + 8 * half;
        float h = acc[r] + bj;
        h = h > 0.0f ? h : 0.0f;
        size_t idx = (size_t)row * HID + j;
        h32[idx] = h;
        h16[idx] = (_Float16)h;
    }
}

// ---------------------------------------------------------------- GRU step
// x == h for t>=1 (reference feeds h_new back as x). FIRST: x==0 -> gi = b_ih.
// grid: (HID/32, BATCH/64), block 256. 6 WMMA accumulators per wave.

template <bool FIRST>
__global__ __launch_bounds__(256) void gru_step(
    const _Float16* __restrict__ h16_prev,  // [BATCH, HID]
    float* __restrict__ h32,                // [BATCH, HID] in-place (elementwise)
    _Float16* __restrict__ h16_next,        // [BATCH, HID]
    const _Float16* __restrict__ Wih,       // [3H, H] f16 row-major
    const _Float16* __restrict__ Whh,       // [3H, H] f16 row-major
    const float* __restrict__ bih,          // [3H]
    const float* __restrict__ bhh) {        // [3H]
    const int lane = threadIdx.x & 31;
    const int w = threadIdx.x >> 5;
    const int wm = w & 3, wn = w >> 2;
    const int m = lane & 15, half = lane >> 4;

    const int row0 = blockIdx.y * 64 + wm * 16;
    const int col0 = blockIdx.x * 32 + wn * 16;
    const int j = col0 + m;  // B-fragment column == output column for this lane

    const _Float16* Arow = h16_prev + (size_t)(row0 + m) * HID;
    const _Float16* Bhr = Whh + (size_t)(0 * HID + j) * HID;
    const _Float16* Bhz = Whh + (size_t)(1 * HID + j) * HID;
    const _Float16* Bhn = Whh + (size_t)(2 * HID + j) * HID;
    const _Float16* Bir = Wih + (size_t)(0 * HID + j) * HID;
    const _Float16* Biz = Wih + (size_t)(1 * HID + j) * HID;
    const _Float16* Bin = Wih + (size_t)(2 * HID + j) * HID;

    v8f a_ir = {}, a_iz = {}, a_in = {};
    v8f a_hr = {}, a_hz = {}, a_hn = {};

    for (int k0 = 0; k0 < HID; k0 += 32) {
        v16h a = load_a16(Arow, k0, half);
        const int kb = k0 + 16 * half;
        v16h b;
        b = LOAD_B(Bhr, kb); WMMA(a_hr, a, b);
        b = LOAD_B(Bhz, kb); WMMA(a_hz, a, b);
        b = LOAD_B(Bhn, kb); WMMA(a_hn, a, b);
        if (!FIRST) {
            b = LOAD_B(Bir, kb); WMMA(a_ir, a, b);
            b = LOAD_B(Biz, kb); WMMA(a_iz, a, b);
            b = LOAD_B(Bin, kb); WMMA(a_in, a, b);
        }
    }

    const float b_ir = bih[j], b_iz = bih[HID + j], b_in = bih[2 * HID + j];
    const float b_hr = bhh[j], b_hz = bhh[HID + j], b_hn = bhh[2 * HID + j];

#pragma unroll
    for (int r = 0; r < 8; ++r) {
        int row = row0 + r + 8 * half;  // C layout: c[r] = C[r + 8*half][lane&15]
        size_t idx = (size_t)row * HID + j;
        float rg = sigmoid_f(a_ir[r] + b_ir + a_hr[r] + b_hr);
        float zg = sigmoid_f(a_iz[r] + b_iz + a_hz[r] + b_hz);
        float hn = a_hn[r] + b_hn;
        float nn = tanh_f(a_in[r] + b_in + rg * hn);
        float hp = h32[idx];
        float h = (1.0f - zg) * nn + zg * hp;
        h32[idx] = h;
        h16_next[idx] = (_Float16)h;
    }
}

// ---------------------------------------------------------------- per-step outputs
// nodes[:, t, :] = h @ Wnode^T + b   (WMMA, 64x128 block tile)
// stop[:, t]     = sigmoid(h . wstop + b)  (4-lane shuffle reduction)
// grid: BATCH/64, block 256

__global__ __launch_bounds__(256) void out_step(
    const _Float16* __restrict__ h16,      // [BATCH, HID] at step t
    const _Float16* __restrict__ Wnode16,  // [FEAT, HID] row-major
    const float* __restrict__ bnode,       // [FEAT]
    const float* __restrict__ wstop,       // [HID]
    const float* __restrict__ bstop,       // [1]
    float* __restrict__ nodes,             // [BATCH, TSTEPS, FEAT]
    float* __restrict__ stop,              // [BATCH, TSTEPS]
    int t) {
    const int lane = threadIdx.x & 31;
    const int w = threadIdx.x >> 5;
    const int wm = w & 3, wn = w >> 2;
    const int m = lane & 15, half = lane >> 4;

    const int row0 = blockIdx.x * 64;
    const int wrow0 = row0 + wm * 16;

    const _Float16* Arow = h16 + (size_t)(wrow0 + m) * HID;
    const _Float16* Brow[4];
    float bb[4];
#pragma unroll
    for (int c = 0; c < 4; ++c) {
        int col = wn * 64 + c * 16 + m;
        Brow[c] = Wnode16 + (size_t)col * HID;
        bb[c] = bnode[col];
    }

    v8f acc[4] = {{}, {}, {}, {}};
    for (int k0 = 0; k0 < HID; k0 += 32) {
        v16h a = load_a16(Arow, k0, half);
        const int kb = k0 + 16 * half;
#pragma unroll
        for (int c = 0; c < 4; ++c) {
            v16h b = LOAD_B(Brow[c], kb);
            WMMA(acc[c], a, b);
        }
    }

#pragma unroll
    for (int c = 0; c < 4; ++c) {
        int col = wn * 64 + c * 16 + m;
#pragma unroll
        for (int r = 0; r < 8; ++r) {
            int row = wrow0 + r + 8 * half;
            nodes[(size_t)row * (TSTEPS * FEAT) + (size_t)t * FEAT + col] =
                acc[c][r] + bb[c];
        }
    }

    // stop probability: 4 threads per row, each covers 128 of 512 elements
    const int row_s = row0 + (threadIdx.x >> 2);
    const int part = threadIdx.x & 3;
    const _Float16* hr = h16 + (size_t)row_s * HID + part * 128;
    const float* wsr = wstop + part * 128;
    float s = 0.0f;
#pragma unroll 4
    for (int k = 0; k < 128; k += 8) {
        v8h hv = *(const v8h*)(hr + k);
#pragma unroll
        for (int q = 0; q < 8; ++q) s += (float)hv[q] * wsr[k + q];
    }
    s += __shfl_xor(s, 1, 32);
    s += __shfl_xor(s, 2, 32);
    if (part == 0)
        stop[(size_t)row_s * TSTEPS + t] = sigmoid_f(s + bstop[0]);
}

// ---------------------------------------------------------------- launcher

extern "C" void kernel_launch(void* const* d_in, const int* in_sizes, int n_in,
                              void* d_out, int out_size, void* d_ws,
                              size_t ws_size, hipStream_t stream) {
    const float* z      = (const float*)d_in[0];
    const float* W_init = (const float*)d_in[1];
    const float* b_init = (const float*)d_in[2];
    const float* W_ih   = (const float*)d_in[3];
    const float* b_ih   = (const float*)d_in[4];
    const float* W_hh   = (const float*)d_in[5];
    const float* b_hh   = (const float*)d_in[6];
    const float* W_node = (const float*)d_in[7];
    const float* b_node = (const float*)d_in[8];
    const float* W_stop = (const float*)d_in[9];
    const float* b_stop = (const float*)d_in[10];

    float* nodes = (float*)d_out;
    float* stop  = nodes + (size_t)BATCH * TSTEPS * FEAT;

    // workspace carve-out (~22 MB), 256-byte aligned sub-buffers
    char* ws = (char*)d_ws;
    size_t off = 0;
    auto take = [&](size_t bytes) -> void* {
        void* p = ws + off;
        off += (bytes + 255) & ~(size_t)255;
        return p;
    };
    _Float16* z16     = (_Float16*)take((size_t)BATCH * ZDIM * 2);
    _Float16* Winit16 = (_Float16*)take((size_t)HID * ZDIM * 2);
    _Float16* Wih16   = (_Float16*)take((size_t)3 * HID * HID * 2);
    _Float16* Whh16   = (_Float16*)take((size_t)3 * HID * HID * 2);
    _Float16* Wnode16 = (_Float16*)take((size_t)FEAT * HID * 2);
    float*    h32     = (float*)take((size_t)BATCH * HID * 4);
    _Float16* h16a    = (_Float16*)take((size_t)BATCH * HID * 2);
    _Float16* h16b    = (_Float16*)take((size_t)BATCH * HID * 2);

    auto cvt = [&](const float* in, _Float16* out, int n) {
        cvt_f32_f16<<<(n + 255) / 256, 256, 0, stream>>>(in, out, n);
    };
    cvt(z, z16, BATCH * ZDIM);
    cvt(W_init, Winit16, HID * ZDIM);
    cvt(W_ih, Wih16, 3 * HID * HID);
    cvt(W_hh, Whh16, 3 * HID * HID);
    cvt(W_node, Wnode16, FEAT * HID);

    dim3 blk(256);
    dim3 grid_gemm(HID / 32, BATCH / 64);
    dim3 grid_out(BATCH / 64);

    init_h0<<<grid_gemm, blk, 0, stream>>>(z16, Winit16, b_init, h32, h16a);

    _Float16* hin = h16a;
    _Float16* hout = h16b;
    for (int t = 0; t < TSTEPS; ++t) {
        if (t == 0)
            gru_step<true><<<grid_gemm, blk, 0, stream>>>(hin, h32, hout, Wih16,
                                                          Whh16, b_ih, b_hh);
        else
            gru_step<false><<<grid_gemm, blk, 0, stream>>>(hin, h32, hout, Wih16,
                                                           Whh16, b_ih, b_hh);
        out_step<<<grid_out, blk, 0, stream>>>(hout, Wnode16, b_node, W_stop,
                                               b_stop, nodes, stop, t);
        _Float16* tmp = hin; hin = hout; hout = tmp;
    }
}
